// CrossAttention_82429012345074
// MI455X (gfx1250) — compile-verified, matrix operand order checked by reading, not run on hardware
//
#include <hip/hip_runtime.h>

#define HEADS   8
#define DHEAD   64
#define NBATCH  2
#define NSEQ    4096
#define DMODEL  512
#define MTOT    (NBATCH * NSEQ)   // 8192 rows

typedef __attribute__((ext_vector_type(16))) _Float16 v16h;
typedef __attribute__((ext_vector_type(8)))  _Float16 v8h;
typedef __attribute__((ext_vector_type(8)))  float    v8f;

// ---------------------------------------------------------------------------
// Fragment loaders per CDNA5 WMMA VGPR layouts (ISA 7.12.2).
// A (16x32 f16, MxK): lane&15 = row M; lanes>=16 hold K=8..15 / 24..31.
// B (32x16 f16, KxN) mirrors with lane&15 = column N over a row-major [N x K]
// source — one helper serves both.
// ---------------------------------------------------------------------------
__device__ __forceinline__ v16h frag_ld_h(const _Float16* p, int ld) {
  const int lane = threadIdx.x & 31;
  const int r    = lane & 15;
  const int kb   = (lane >> 4) << 3;       // 0 or 8
  const _Float16* q = p + r * ld + kb;
  v8h lo = *(const v8h*)(q);
  v8h hi = *(const v8h*)(q + 16);
  v16h a;
#pragma unroll
  for (int j = 0; j < 8; ++j) { a[j] = lo[j]; a[j + 8] = hi[j]; }
  return a;
}

__device__ __forceinline__ float redmax16(float v) {
#pragma unroll
  for (int xm = 1; xm < 16; xm <<= 1) v = fmaxf(v, __shfl_xor(v, xm, 32));
  return v;
}
__device__ __forceinline__ float redsum16(float v) {
#pragma unroll
  for (int xm = 1; xm < 16; xm <<= 1) v += __shfl_xor(v, xm, 32);
  return v;
}

// 16-byte global -> LDS async copy (GLOBAL_LOAD_ASYNC_TO_LDS_B128, ASYNCcnt).
// VDST = per-lane LDS byte address (low 32 bits of a generic shared pointer
// are the LDS offset per the aperture rules); VADDR = 64-bit global address.
__device__ __forceinline__ void cp16_g2l(void* lptr, const void* gptr) {
  const uint32_t loff  = (uint32_t)(uintptr_t)lptr;
  const uint64_t gaddr = (uint64_t)(uintptr_t)gptr;
  asm volatile("global_load_async_to_lds_b128 %0, %1, off"
               :
               : "v"(loff), "v"(gaddr)
               : "memory");
}

__device__ __forceinline__ void wait_async_cp() {
  asm volatile("s_wait_asynccnt 0x0" ::: "memory");
}

// ---------------------------------------------------------------------------
// Kernel 0: fp32 -> f16 conversion (n must be a multiple of 2048).
// ---------------------------------------------------------------------------
__global__ __launch_bounds__(256) void cvt_f16_kernel(
    const float* __restrict__ s, _Float16* __restrict__ d) {
  const size_t i = ((size_t)blockIdx.x * 256 + threadIdx.x) * 8;
#pragma unroll
  for (int j = 0; j < 8; ++j) d[i + j] = (_Float16)s[i + j];
}

// ---------------------------------------------------------------------------
// Kernel 1: fused QKV projection from pre-converted f16 x / W.
// Grid (MTOT/16, 3), 256 threads = 8 waves; wave w owns cols [64w, 64w+64).
// Output f16, head-major [b, h, n, d].
// ---------------------------------------------------------------------------
__global__ __launch_bounds__(256) void qkv_proj_kernel(
    const _Float16* __restrict__ xh,
    const _Float16* __restrict__ wqh, const _Float16* __restrict__ wkh,
    const _Float16* __restrict__ wvh,
    _Float16* __restrict__ qws, _Float16* __restrict__ kws,
    _Float16* __restrict__ vws) {
  const int wave = threadIdx.x >> 5;
  const int lane = threadIdx.x & 31;
  const int m0   = blockIdx.x * 16;
  const int n0   = wave * 64;

  const _Float16* W = (blockIdx.y == 0) ? wqh : (blockIdx.y == 1) ? wkh : wvh;
  _Float16* dst     = (blockIdx.y == 0) ? qws : (blockIdx.y == 1) ? kws : vws;

  v8f zf = {};
  v8f acc[4];
#pragma unroll
  for (int t = 0; t < 4; ++t) acc[t] = zf;

  for (int ks = 0; ks < DMODEL; ks += 32) {
    v16h a = frag_ld_h(xh + (size_t)m0 * DMODEL + ks, DMODEL);
#pragma unroll
    for (int t = 0; t < 4; ++t) {
      v16h b = frag_ld_h(W + (size_t)(n0 + t * 16) * DMODEL + ks, DMODEL);
      acc[t] = __builtin_amdgcn_wmma_f32_16x16x32_f16(
          false, a, false, b, (short)0, acc[t], false, false);
    }
  }

  const int hi = lane >> 4;
#pragma unroll
  for (int t = 0; t < 4; ++t) {
    const int n = n0 + t * 16 + (lane & 15);
    const int h = n >> 6, dd = n & 63;
#pragma unroll
    for (int r = 0; r < 8; ++r) {
      const int m = m0 + r + 8 * hi;            // global row in [B*N]
      const int b = m >> 12, i = m & (NSEQ - 1);
      dst[((size_t)((b * HEADS + h) * NSEQ) + i) * DHEAD + dd] =
          (_Float16)acc[t][r];
    }
  }
}

// ---------------------------------------------------------------------------
// Kernel 2: flash attention.  Grid (NSEQ/128, B*H), 256 threads = 8 waves.
// Block owns 128 query rows of one (b,h); wave owns 16 rows.
// K tile staged via async global->LDS; V staged transposed; online softmax.
// ---------------------------------------------------------------------------
__global__ __launch_bounds__(256) void flash_attn_kernel(
    const _Float16* __restrict__ qws, const _Float16* __restrict__ kws,
    const _Float16* __restrict__ vws, _Float16* __restrict__ ows) {
  __shared__ _Float16 lds_k[128 * 64];        // K tile, row-major [j][d]
  __shared__ _Float16 lds_vt[64 * 128];       // V tile transposed [d][j]
  __shared__ _Float16 lds_p[8 * 16 * 128];    // per-wave P scratch

  const int tid  = threadIdx.x;
  const int wave = tid >> 5;
  const int lane = tid & 31;
  const int hi   = lane >> 4;
  const int bh   = blockIdx.y;                // 0..15
  const int b    = bh >> 3, h = bh & 7;
  const int i0   = blockIdx.x * 128;

  const size_t head_off = (size_t)bh * NSEQ * DHEAD;

  // Q fragments for this wave's 16 rows, scale = 1/sqrt(64) folded in (exact).
  v16h qf[2];
  {
    const _Float16* qb = qws + head_off + (size_t)(i0 + wave * 16) * DHEAD;
#pragma unroll
    for (int kc = 0; kc < 2; ++kc) {
      qf[kc] = frag_ld_h(qb + kc * 32, DHEAD);
#pragma unroll
      for (int j = 0; j < 16; ++j) qf[kc][j] = qf[kc][j] * (_Float16)0.125f;
    }
  }

  v8f zf = {};
  v8f O[4];
#pragma unroll
  for (int t = 0; t < 4; ++t) O[t] = zf;
  float mrun[8], lrun[8];
#pragma unroll
  for (int r = 0; r < 8; ++r) { mrun[r] = -1e30f; lrun[r] = 0.f; }

  for (int j0 = 0; j0 < NSEQ; j0 += 128) {
    // ---- stage K (async global->LDS) and V (transposed) ----
    {
      const char* gk = (const char*)(kws + head_off + (size_t)j0 * DHEAD);
      char* sk = (char*)lds_k;
      for (int c = tid; c < 1024; c += 256)
        cp16_g2l(sk + (size_t)c * 16, gk + (size_t)c * 16);
      const _Float16* gv = vws + head_off + (size_t)j0 * DHEAD;
      for (int c = tid; c < 8192; c += 256) {
        int r = c >> 6, d = c & 63;
        lds_vt[d * 128 + r] = gv[c];
      }
      // prefetch next tiles into cache while this tile computes
      if (j0 + 128 < NSEQ) {
        __builtin_prefetch(kws + head_off + (size_t)(j0 + 128) * DHEAD +
                               (size_t)tid * 32, 0, 1);
        __builtin_prefetch(vws + head_off + (size_t)(j0 + 128) * DHEAD +
                               (size_t)tid * 32, 0, 1);
      }
      wait_async_cp();
    }
    __syncthreads();

    // ---- S = (Q*scale) @ K^T : 16 x 128 in 8 accumulator tiles ----
    v8f S[8];
#pragma unroll
    for (int t = 0; t < 8; ++t) S[t] = zf;
#pragma unroll
    for (int kc = 0; kc < 2; ++kc) {
#pragma unroll
      for (int t = 0; t < 8; ++t) {
        v16h bk = frag_ld_h(&lds_k[(t * 16) * DHEAD + kc * 32], DHEAD);
        S[t] = __builtin_amdgcn_wmma_f32_16x16x32_f16(
            false, qf[kc], false, bk, (short)0, S[t], false, false);
      }
    }

    // ---- online softmax (row stats via 16-lane shuffle reductions) ----
    _Float16* pw = lds_p + wave * (16 * 128);
#pragma unroll
    for (int r = 0; r < 8; ++r) {
      float mx = -1e30f;
#pragma unroll
      for (int t = 0; t < 8; ++t) mx = fmaxf(mx, S[t][r]);
      mx = redmax16(mx);
      const float mnew  = fmaxf(mrun[r], mx);
      const float alpha = __expf(mrun[r] - mnew);
      float s = 0.f;
#pragma unroll
      for (int t = 0; t < 8; ++t) {
        float p = __expf(S[t][r] - mnew);
        s += p;
        pw[(r + 8 * hi) * 128 + t * 16 + (lane & 15)] = (_Float16)p;
      }
      s = redsum16(s);
      lrun[r] = lrun[r] * alpha + s;
      mrun[r] = mnew;
#pragma unroll
      for (int t = 0; t < 4; ++t) O[t][r] *= alpha;
    }

    // ---- O += P @ V ----
#pragma unroll
    for (int kc = 0; kc < 4; ++kc) {
      v16h ap = frag_ld_h(pw + kc * 32, 128);
#pragma unroll
      for (int t = 0; t < 4; ++t) {
        v16h bv = frag_ld_h(&lds_vt[(t * 16) * 128 + kc * 32], 128);
        O[t] = __builtin_amdgcn_wmma_f32_16x16x32_f16(
            false, ap, false, bv, (short)0, O[t], false, false);
      }
    }
    __syncthreads();
  }

  // ---- normalize, store f16 token-major [b, n, h*d] ----
#pragma unroll
  for (int r = 0; r < 8; ++r) {
    const float inv = 1.f / lrun[r];
    const int row = i0 + wave * 16 + r + 8 * hi;
#pragma unroll
    for (int t = 0; t < 4; ++t) {
      const int d = t * 16 + (lane & 15);
      ows[((size_t)(b * NSEQ + row)) * DMODEL + h * DHEAD + d] =
          (_Float16)(O[t][r] * inv);
    }
  }
}

// ---------------------------------------------------------------------------
// Kernel 3: output projection  out = o @ Wo^T + bo  (fp32 result).
// ---------------------------------------------------------------------------
__global__ __launch_bounds__(256) void out_proj_kernel(
    const _Float16* __restrict__ ows, const _Float16* __restrict__ woh,
    const float* __restrict__ bo, float* __restrict__ out) {
  const int wave = threadIdx.x >> 5;
  const int lane = threadIdx.x & 31;
  const int m0   = blockIdx.x * 16;
  const int n0   = wave * 64;

  v8f zf = {};
  v8f acc[4];
#pragma unroll
  for (int t = 0; t < 4; ++t) acc[t] = zf;

  for (int ks = 0; ks < DMODEL; ks += 32) {
    v16h a = frag_ld_h(ows + (size_t)m0 * DMODEL + ks, DMODEL);
#pragma unroll
    for (int t = 0; t < 4; ++t) {
      v16h b = frag_ld_h(woh + (size_t)(n0 + t * 16) * DMODEL + ks, DMODEL);
      acc[t] = __builtin_amdgcn_wmma_f32_16x16x32_f16(
          false, a, false, b, (short)0, acc[t], false, false);
    }
  }

  const int hi = lane >> 4;
#pragma unroll
  for (int t = 0; t < 4; ++t) {
    const int n = n0 + t * 16 + (lane & 15);
    const float bias = bo[n];
#pragma unroll
    for (int r = 0; r < 8; ++r) {
      const int m = m0 + r + 8 * hi;
      out[(size_t)m * DMODEL + n] = acc[t][r] + bias;
    }
  }
}

// ---------------------------------------------------------------------------
extern "C" void kernel_launch(void* const* d_in, const int* in_sizes, int n_in,
                              void* d_out, int out_size, void* d_ws,
                              size_t ws_size, hipStream_t stream) {
  (void)in_sizes; (void)n_in; (void)out_size; (void)ws_size;
  const float* x  = (const float*)d_in[0];
  const float* Wq = (const float*)d_in[1];
  const float* Wk = (const float*)d_in[2];
  const float* Wv = (const float*)d_in[3];
  const float* Wo = (const float*)d_in[4];
  const float* bo = (const float*)d_in[5];
  float* out = (float*)d_out;

  // f16 workspace layout
  const size_t EX = (size_t)MTOT * DMODEL;                 // 4,194,304 (x, o)
  const size_t EW = (size_t)DMODEL * DMODEL;               // 262,144 per W
  const size_t EH = (size_t)NBATCH * HEADS * NSEQ * DHEAD; // 4,194,304 (q/k/v)
  _Float16* xh  = (_Float16*)d_ws;
  _Float16* wqh = xh + EX;
  _Float16* wkh = wqh + EW;
  _Float16* wvh = wkh + EW;
  _Float16* woh = wvh + EW;
  _Float16* qws = woh + EW;
  _Float16* kws = qws + EH;
  _Float16* vws = kws + EH;
  _Float16* ows = vws + EH;

  cvt_f16_kernel<<<dim3(EX / 2048), 256, 0, stream>>>(x, xh);
  cvt_f16_kernel<<<dim3(EW / 2048), 256, 0, stream>>>(Wq, wqh);
  cvt_f16_kernel<<<dim3(EW / 2048), 256, 0, stream>>>(Wk, wkh);
  cvt_f16_kernel<<<dim3(EW / 2048), 256, 0, stream>>>(Wv, wvh);
  cvt_f16_kernel<<<dim3(EW / 2048), 256, 0, stream>>>(Wo, woh);

  qkv_proj_kernel<<<dim3(MTOT / 16, 3), 256, 0, stream>>>(xh, wqh, wkh, wvh,
                                                          qws, kws, vws);
  flash_attn_kernel<<<dim3(NSEQ / 128, NBATCH * HEADS), 256, 0, stream>>>(
      qws, kws, vws, ows);
  out_proj_kernel<<<dim3(MTOT / 16), 256, 0, stream>>>(ows, woh, bo, out);
}
